// Surf_10325101379707
// MI455X (gfx1250) — matrix operand order repaired and checked by reading, not compiled, not run on hardware
//
#include <hip/hip_runtime.h>

typedef __attribute__((ext_vector_type(16))) _Float16 v16h;
typedef __attribute__((ext_vector_type(8)))  _Float16 v8h;
typedef __attribute__((ext_vector_type(8)))  float    v8f;
typedef __attribute__((ext_vector_type(4)))  unsigned int u32x4;
typedef __attribute__((ext_vector_type(8)))  int      i32x8;
typedef __attribute__((ext_vector_type(4)))  int      i32x4;

#define L_LIGHTS 8
#define P_PTS    512
#define HID      128
#define NC       64
#define NF       128
#define NA       192          // NC + NF
#define NRAYS    (L_LIGHTS * P_PTS)   // 4096
#define S_SCALE  100.0f
#define LV_NEAR  1e-3f
#define LV_FAR   0.5f

#define MLP_ROWS    128       // rows (sample points) per workgroup
#define MLP_THREADS 128       // 4 waves of 32; each wave owns 32 rows
#define WAVE_ROWS   32

__device__ __forceinline__ float sigmoid_s(float x) {
    return 1.0f / (1.0f + __expf(-x * S_SCALE));
}
__device__ __forceinline__ float coarse_d(float i) {
    float t = i * (1.0f / 64.0f);
    return LV_NEAR * (1.0f - t) + LV_FAR * t;
}

// ---------------------------------------------------------------------------
// Transpose w1,w2 to f16 column-major copies: wt16[layer][n*128 + k] = W[k][n]
// ---------------------------------------------------------------------------
__global__ void prep_weights(const float* __restrict__ w1,
                             const float* __restrict__ w2,
                             _Float16* __restrict__ wt16) {
    int idx = blockIdx.x * 256 + threadIdx.x;          // 0 .. 32767
    int layer = idx >> 14;
    int r = idx & (HID * HID - 1);
    int n = r >> 7, k = r & 127;
    const float* W = layer ? w2 : w1;
    wt16[idx] = (_Float16)W[k * HID + n];
}

// ---------------------------------------------------------------------------
// Batched SDF MLP: 3 -> 128 -> 128 -> 128 -> 1.
// Hidden layers via V_WMMA_F32_16X16X32_F16. One workgroup = 128 rows,
// 4 waves x 32 rows: two A-tiles per wave -> each B fragment feeds 2 WMMAs.
// Both layers' weights (64 KB f16) are DMA'd into LDS by the Tensor Data
// Mover, overlapped with position setup and the VALU input layer.
// ---------------------------------------------------------------------------
template<int NSAMP, bool USE_DALL>
__global__ __launch_bounds__(MLP_THREADS) void mlp_kernel(
    const float* __restrict__ pts,          // [P,3]
    const float* __restrict__ light,        // [L,3]
    const float* __restrict__ dall,         // [NRAYS,192] (pass 2) or unused
    const float* __restrict__ w0, const float* __restrict__ b0,
    const float* __restrict__ b1, const float* __restrict__ b2,
    const float* __restrict__ w3, const float* __restrict__ b3,
    const _Float16* __restrict__ wt16,      // [2][128*128] transposed f16
    float* __restrict__ sdf_out)            // [NRAYS*NSAMP]
{
    __shared__ __align__(16) _Float16 sW[2 * HID * HID];   // 64 KB, wT[n][k] x2
    __shared__ __align__(16) _Float16 sH[MLP_ROWS * HID];  // 32 KB activations
    __shared__ float sBias[2][HID];
    __shared__ float sW3[HID];
    __shared__ float sPos[MLP_ROWS][4];

    const int tid  = threadIdx.x;
    const int lane = tid & 31;
    const int wave = tid >> 5;                     // 0..3
    const long long rowBase = (long long)blockIdx.x * MLP_ROWS;

    // ---- kick off TDM: DMA both weight matrices (64 KB) into LDS ----------
    if (tid == 0) {
        const unsigned td0 = (2 * HID * HID) / 4;  // 8192 8-byte elements
        unsigned lds_addr = (unsigned)(__SIZE_TYPE__)&sW[0];
        unsigned long long ga = (unsigned long long)(__SIZE_TYPE__)wt16;
        u32x4 g0;
        g0[0] = 1u;                                           // count=1, user D#
        g0[1] = lds_addr;                                     // lds_addr[63:32]
        g0[2] = (unsigned)(ga & 0xFFFFFFFFu);                 // global_addr lo
        g0[3] = (unsigned)((ga >> 32) & 0x1FFFFFFu)           // global_addr hi
              | 0x80000000u;                                  // type=2 ("image")
        i32x8 g1;
        g1[0] = (3 << 16);                 // wg_mask=0, data_size=3 (8 B)
        g1[1] = (int)((td0 & 0xFFFFu) << 16);       // tensor_dim0[15:0]
        g1[2] = (int)((td0 >> 16) | (1u << 16));    // tensor_dim0[31:16] | tensor_dim1=1
        g1[3] = (int)(td0 << 16);                   // tile_dim0 = 8192
        g1[4] = 1;                                  // tile_dim1 = 1
        g1[5] = (int)td0;                           // tensor_dim0_stride
        g1[6] = 0;
        g1[7] = 0;
        i32x4 gz4 = {0, 0, 0, 0};
        i32x8 gz8 = {0, 0, 0, 0, 0, 0, 0, 0};
        __builtin_amdgcn_tensor_load_to_lds(g0, g1, gz4, gz4, gz8, 0);
    }

    // ---- sample positions: pt + d * normalize(light - pt) (overlaps DMA) ---
    {
        long long row = rowBase + tid;
        int s        = (int)(row % NSAMP);
        long long ray = row / NSAMP;
        int p = (int)(ray % P_PTS);
        int l = (int)(ray / P_PTS);
        float px = pts[p * 3 + 0], py = pts[p * 3 + 1], pz = pts[p * 3 + 2];
        float dx = light[l * 3 + 0] - px;
        float dy = light[l * 3 + 1] - py;
        float dz = light[l * 3 + 2] - pz;
        float inv = rsqrtf(dx * dx + dy * dy + dz * dz);
        dx *= inv; dy *= inv; dz *= inv;
        float d;
        if (USE_DALL) d = dall[row];               // row == ray*192 + s
        else          d = coarse_d((float)s);
        sPos[tid][0] = px + d * dx;
        sPos[tid][1] = py + d * dy;
        sPos[tid][2] = pz + d * dz;
    }
    if (tid < HID) {
        sW3[tid]      = w3[tid];
        sBias[0][tid] = b1[tid];
        sBias[1][tid] = b2[tid];
    }
    __syncthreads();   // positions visible

    // ---- layer 0: 3 -> 128 (VALU, overlaps DMA) ----
    for (int idx = tid; idx < MLP_ROWS * HID; idx += MLP_THREADS) {
        int r = idx >> 7;
        int c = idx & 127;
        float h = b0[c] + sPos[r][0] * w0[c]
                        + sPos[r][1] * w0[HID + c]
                        + sPos[r][2] * w0[2 * HID + c];
        sH[r * HID + c] = (_Float16)fmaxf(h, 0.0f);
    }

    // ---- wait for the TDM transfer, then make everything visible ----
    if (tid == 0) __builtin_amdgcn_s_wait_tensorcnt(0);
    __syncthreads();

    // per-lane WMMA fragment geometry (ISA 16-bit A / B layouts, wave32)
    const int wrow0   = wave * WAVE_ROWS;          // first row of this wave
    const int arow0   = wrow0 + (lane & 15);       // A row, tile 0
    const int arow1   = arow0 + 16;                // A row, tile 1
    const int aKsel   = (lane >> 4) * 8;           // K sub-chunk select
    const int bcol    = lane & 15;                 // B column within tile
    const int bKsel   = (lane >> 4) * 8;
    const int orow0   = wrow0 + (lane >> 4) * 8;   // C/D row base, tile 0
    const int orow1   = orow0 + 16;                // C/D row base, tile 1
    const int ocol    = lane & 15;

    // ---- layers 1,2: 128 -> 128 via WMMA ----
    for (int layer = 0; layer < 2; ++layer) {
        const _Float16* sWL = &sW[layer * HID * HID];

        // pre-load this wave's 32x128 A block (8 fragments of 16x32)
        v16h afrag0[4], afrag1[4];
        #pragma unroll
        for (int kt = 0; kt < 4; ++kt) {
            const _Float16* ap0 = &sH[arow0 * HID + kt * 32 + aKsel];
            v8h lo0 = *(const v8h*)ap0;
            v8h hi0 = *(const v8h*)(ap0 + 16);
            afrag0[kt] = __builtin_shufflevector(lo0, hi0,
                0, 1, 2, 3, 4, 5, 6, 7, 8, 9, 10, 11, 12, 13, 14, 15);
            const _Float16* ap1 = &sH[arow1 * HID + kt * 32 + aKsel];
            v8h lo1 = *(const v8h*)ap1;
            v8h hi1 = *(const v8h*)(ap1 + 16);
            afrag1[kt] = __builtin_shufflevector(lo1, hi1,
                0, 1, 2, 3, 4, 5, 6, 7, 8, 9, 10, 11, 12, 13, 14, 15);
        }

        // 8 column tiles of 16; each B fragment feeds two WMMAs (M reuse)
        for (int ct = 0; ct < 8; ++ct) {
            const int cb = ct * 16;
            v8f acc0 = {};
            v8f acc1 = {};
            #pragma unroll
            for (int kt = 0; kt < 4; ++kt) {
                const _Float16* bp = &sWL[(cb + bcol) * HID + kt * 32 + bKsel];
                v8h lo = *(const v8h*)bp;
                v8h hi = *(const v8h*)(bp + 16);
                v16h bfrag = __builtin_shufflevector(lo, hi,
                    0, 1, 2, 3, 4, 5, 6, 7, 8, 9, 10, 11, 12, 13, 14, 15);
                acc0 = __builtin_amdgcn_wmma_f32_16x16x32_f16(
                    false, afrag0[kt], false, bfrag, (short)0, acc0, false, false);
                acc1 = __builtin_amdgcn_wmma_f32_16x16x32_f16(
                    false, afrag1[kt], false, bfrag, (short)0, acc1, false, false);
            }
            // bias + ReLU + f16 store (own rows only -> no cross-wave hazard)
            float bias = sBias[layer][cb + ocol];
            #pragma unroll
            for (int r = 0; r < 8; ++r) {
                float v0 = fmaxf(acc0[r] + bias, 0.0f);
                sH[(orow0 + r) * HID + (cb + ocol)] = (_Float16)v0;
                float v1 = fmaxf(acc1[r] + bias, 0.0f);
                sH[(orow1 + r) * HID + (cb + ocol)] = (_Float16)v1;
            }
        }
        __syncthreads();   // rows complete before next layer / final read
    }

    // ---- layer 3: 128 -> 1 (VALU dot, vectorized LDS reads) ----
    {
        float acc = b3[0];
        const v8h* hp = (const v8h*)&sH[tid * HID];
        #pragma unroll
        for (int kv = 0; kv < HID / 8; ++kv) {
            v8h hv = hp[kv];
            #pragma unroll
            for (int e = 0; e < 8; ++e)
                acc += (float)hv[e] * sW3[kv * 8 + e];
        }
        sdf_out[rowBase + tid] = acc;
    }
}

// ---------------------------------------------------------------------------
// Per-ray: sdf -> weights -> cdf -> importance sample 128 fine depths ->
// merge with 64 coarse depths -> sort 192 -> d_all.
// One thread per ray, 32 threads/block, conflict-free stride-33 LDS slices.
// ---------------------------------------------------------------------------
__global__ __launch_bounds__(32) void resample_kernel(
    const float* __restrict__ sdfc,     // [NRAYS*64]
    const float* __restrict__ u,        // [NRAYS*128]
    float* __restrict__ dall_out)       // [NRAYS*192]
{
    __shared__ float sCdf[NC * 33];     // cdf[64] per thread
    __shared__ float sD[NA * 33];       // merged depths per thread

    const int t = threadIdx.x;
    const long long ray = (long long)blockIdx.x * 32 + t;
    const float* sc = sdfc + ray * NC;

    // weights + unnormalized cumsum (cdf)
    float sig_prev = sigmoid_s(sc[0]);
    float T = 1.0f, csum = 0.0f;
    sCdf[t] = 0.0f;
    for (int i = 0; i < NC - 1; ++i) {
        float sig_next = sigmoid_s(sc[i + 1]);
        float alpha = fmaxf((sig_prev - sig_next) / (sig_prev + 1e-10f), 0.0f);
        float w = alpha * T + 1e-5f;            // sample_pdf adds 1e-5
        T *= (1.0f - alpha + 1e-10f);
        csum += w;
        sCdf[(i + 1) * 33 + t] = csum;
        sig_prev = sig_next;
    }
    float invs = 1.0f / csum;
    for (int i = 1; i < NC; ++i) sCdf[i * 33 + t] *= invs;

    // coarse depths
    for (int i = 0; i < NC; ++i) sD[i * 33 + t] = coarse_d((float)i);

    // inverse-CDF fine sampling
    const float* uu = u + ray * NF;
    for (int j = 0; j < NF; ++j) {
        float uv = uu[j];
        int inds = 0;
        for (int i = 0; i < NC; ++i) inds += (uv >= sCdf[i * 33 + t]) ? 1 : 0;
        int below = min(max(inds - 1, 0), NC - 1);
        int above = min(inds, NC - 1);
        float cb = sCdf[below * 33 + t];
        float ca = sCdf[above * 33 + t];
        float bb = coarse_d((float)below);
        float ba = coarse_d((float)above);
        float denom = ca - cb;
        if (denom < 1e-5f) denom = 1.0f;
        float tt = (uv - cb) / denom;
        sD[(NC + j) * 33 + t] = bb + tt * (ba - bb);
    }

    // insertion sort the 192 depths
    for (int i = 1; i < NA; ++i) {
        float key = sD[i * 33 + t];
        int j = i - 1;
        while (j >= 0 && sD[j * 33 + t] > key) {
            sD[(j + 1) * 33 + t] = sD[j * 33 + t];
            --j;
        }
        sD[(j + 1) * 33 + t] = key;
    }

    float* od = dall_out + ray * NA;
    for (int i = 0; i < NA; ++i) od[i] = sD[i * 33 + t];
}

// ---------------------------------------------------------------------------
// Per-ray: sdf_all -> weights -> occlusion -> 1 - sum(w)
// ---------------------------------------------------------------------------
__global__ __launch_bounds__(256) void finalize_kernel(
    const float* __restrict__ sdfa,     // [NRAYS*192]
    float* __restrict__ out)            // [NRAYS]
{
    int ray = blockIdx.x * blockDim.x + threadIdx.x;
    if (ray >= NRAYS) return;
    const float* s = sdfa + (long long)ray * NA;
    float sig_prev = sigmoid_s(s[0]);
    float T = 1.0f, occ = 0.0f;
    for (int i = 0; i < NA - 1; ++i) {
        float sig_next = sigmoid_s(s[i + 1]);
        float alpha = fmaxf((sig_prev - sig_next) / (sig_prev + 1e-10f), 0.0f);
        occ += alpha * T;
        T *= (1.0f - alpha + 1e-10f);
        sig_prev = sig_next;
    }
    out[ray] = 1.0f - occ;
}

// ---------------------------------------------------------------------------
extern "C" void kernel_launch(void* const* d_in, const int* in_sizes, int n_in,
                              void* d_out, int out_size, void* d_ws, size_t ws_size,
                              hipStream_t stream) {
    const float* pts   = (const float*)d_in[0];   // [1,P,3]
    const float* light = (const float*)d_in[1];   // [L,3]
    const float* u     = (const float*)d_in[2];   // [L,P,128]
    const float* w0    = (const float*)d_in[3];
    const float* b0    = (const float*)d_in[4];
    const float* w1    = (const float*)d_in[5];
    const float* b1    = (const float*)d_in[6];
    const float* w2    = (const float*)d_in[7];
    const float* b2    = (const float*)d_in[8];
    const float* w3    = (const float*)d_in[9];
    const float* b3    = (const float*)d_in[10];
    float* out = (float*)d_out;

    // workspace carve-up
    char* ws = (char*)d_ws;
    _Float16* wt16 = (_Float16*)ws;                       ws += 2 * HID * HID * sizeof(_Float16); // 64 KB
    float* sdfc = (float*)ws;                             ws += (size_t)NRAYS * NC * sizeof(float); // 1 MB
    float* dall = (float*)ws;                             ws += (size_t)NRAYS * NA * sizeof(float); // 3 MB
    float* sdfa = (float*)ws;                             // 3 MB

    // 1) transpose + f16-convert hidden-layer weights
    prep_weights<<<(2 * HID * HID) / 256, 256, 0, stream>>>(w1, w2, wt16);

    // 2) coarse SDF: 8*512*64 = 262144 rows
    {
        int nblk = (NRAYS * NC) / MLP_ROWS;               // 2048
        mlp_kernel<NC, false><<<nblk, MLP_THREADS, 0, stream>>>(
            pts, light, nullptr, w0, b0, b1, b2, w3, b3, wt16, sdfc);
    }

    // 3) importance resampling + merge-sort of depths
    resample_kernel<<<NRAYS / 32, 32, 0, stream>>>(sdfc, u, dall);

    // 4) fine SDF: 8*512*192 = 786432 rows
    {
        int nblk = (NRAYS * NA) / MLP_ROWS;               // 6144
        mlp_kernel<NA, true><<<nblk, MLP_THREADS, 0, stream>>>(
            pts, light, dall, w0, b0, b1, b2, w3, b3, wt16, sdfa);
    }

    // 5) visibility
    finalize_kernel<<<(NRAYS + 255) / 256, 256, 0, stream>>>(sdfa, out);
}
